// DeformDFN1_42442866819724
// MI455X (gfx1250) — matrix operand order repaired and checked by reading, not compiled, not run on hardware
//
#include <hip/hip_runtime.h>
#include <hip/hip_bf16.h>
#include <stdint.h>

typedef __attribute__((ext_vector_type(16))) _Float16 v16h;
typedef __attribute__((ext_vector_type(8)))  float    v8f;

// ---------------------------------------------------------------------------
// utility kernels
// ---------------------------------------------------------------------------
__global__ void zero_f32(float* __restrict__ p, int n) {
    int t = blockIdx.x * blockDim.x + threadIdx.x;
    if (t < n) p[t] = 0.0f;
}

// Gather frame for this step into contiguous (B,1,64,64).
__global__ void pack_frame(const float* __restrict__ in10,
                           const float* __restrict__ prev,
                           float* __restrict__ frame, int step, int n) {
    int t = blockIdx.x * blockDim.x + threadIdx.x;
    if (t < n) frame[t] = (step < 10) ? in10[(size_t)t * 10 + step] : prev[t];
}

// ---------------------------------------------------------------------------
// Pack weights straight into WMMA B-fragment layout:
//   wp[((kc*(Np/16)+nt)*32 + lane)*16 + e]  <->  K = kc*32+(lane>>4)*8+e+(e>=8?8:0),
//                                                N = nt*16+(lane&15)
// conv  : w (Cout,Cin,KH,KH), K index = (ci*KH+ky)*KH+kx
// deconv: w (Cin,Cout,3,3),   wt[n,ci,ky,kx] = w[ci,n,2-ky,2-kx]
// ---------------------------------------------------------------------------
__global__ void pack_w_frag(const float* __restrict__ w, _Float16* __restrict__ wp,
                            int Kc, int Np, int Cin, int Cout, int KH,
                            int Ktot, int deconv) {
    int t = blockIdx.x * blockDim.x + threadIdx.x;
    int total = Kc * (Np >> 4) * 32;
    if (t >= total) return;
    int lane = t & 31;
    int rest = t >> 5;
    int nt = rest % (Np >> 4);
    int kc = rest / (Np >> 4);
    int n = nt * 16 + (lane & 15);
    int kkh = KH * KH;
    v16h hv;
#pragma unroll
    for (int e = 0; e < 16; ++e) {
        int k = kc * 32 + (lane >> 4) * 8 + e + ((e >= 8) ? 8 : 0);
        float v = 0.0f;
        if (k < Ktot && n < Cout) {
            if (!deconv) {
                int ci = k / kkh, r = k % kkh, ky = r / KH, kx = r % KH;
                v = w[(((size_t)n * Cin + ci) * KH + ky) * KH + kx];
            } else {
                int ci = k / 9, r = k % 9, ky = r / 3, kx = r % 3;
                v = w[(((size_t)ci * Cout + n) * 3 + (2 - ky)) * 3 + (2 - kx)];
            }
        }
        hv[e] = (_Float16)v;
    }
    *(v16h*)(wp + (size_t)t * 16) = hv;
}

// ---------------------------------------------------------------------------
// Build im2col A matrix in WMMA A-fragment layout:
//   A[((mt*Kc + kc)*32 + lane)*16 + e] <-> M = mt*16+(lane&15),
//                                          K = kc*32+(lane>>4)*8+e+(e>=8?8:0)
// deconv=1: gather from 2x lhs-dilated, pad-1 input.
// ---------------------------------------------------------------------------
__global__ void build_a(const float* __restrict__ in, _Float16* __restrict__ A,
                        int Mtiles, int Kc, int B, int Cin, int IH, int IW,
                        int OH, int OW, int KH, int stride, int pad,
                        int Ktot, int deconv) {
    int t = blockIdx.x * blockDim.x + threadIdx.x;
    int total = Mtiles * Kc * 32;
    if (t >= total) return;
    int lane = t & 31;
    int rest = t >> 5;
    int kc = rest % Kc;
    int mt = rest / Kc;
    int m = mt * 16 + (lane & 15);
    int Mtot = B * OH * OW;
    bool mv = (m < Mtot);
    int bI = 0, oy = 0, ox = 0;
    if (mv) { bI = m / (OH * OW); int r = m % (OH * OW); oy = r / OW; ox = r % OW; }
    int kkh = KH * KH;
    v16h hv;
#pragma unroll
    for (int e = 0; e < 16; ++e) {
        int k = kc * 32 + (lane >> 4) * 8 + e + ((e >= 8) ? 8 : 0);
        float a = 0.0f;
        if (mv && k < Ktot) {
            int ci = k / kkh, r = k % kkh, ky = r / KH, kx = r % KH;
            if (!deconv) {
                int iy = oy * stride - pad + ky;
                int ix = ox * stride - pad + kx;
                if (iy >= 0 && iy < IH && ix >= 0 && ix < IW)
                    a = in[(((size_t)bI * Cin + ci) * IH + iy) * IW + ix];
            } else {
                int py = oy + ky - 1, px = ox + kx - 1;
                if (py >= 0 && px >= 0 && ((py | px) & 1) == 0) {
                    int iy = py >> 1, ix = px >> 1;
                    if (iy < IH && ix < IW)
                        a = in[(((size_t)bI * Cin + ci) * IH + iy) * IW + ix];
                }
            }
        }
        hv[e] = (_Float16)a;
    }
    *(v16h*)(A + (size_t)t * 16) = hv;
}

// ---------------------------------------------------------------------------
// Fragment GEMM: one wave owns a 16x(16*NT) output tile.
// Per K-chunk: 1 A-fragment vector load + NT B-fragment loads + NT WMMAs.
// Epilogue: +bias, leaky(0.01), *g*rsqrt(1+eps)+bb, optional residual.
// ---------------------------------------------------------------------------
template <int NT>
__global__ void __launch_bounds__(32) wmma_gemm(
    const _Float16* __restrict__ A, const _Float16* __restrict__ Bp,
    const float* __restrict__ bias, const float* __restrict__ gsc,
    const float* __restrict__ bsh, const float* __restrict__ residual,
    float* __restrict__ out,
    int Kc, int Np, int Mtot, int Cout, int OH, int OW)
{
    const int lane = threadIdx.x;
    const int hsel = lane >> 4;
    const int lr   = lane & 15;
    const int mBase = blockIdx.x * 16;
    const int ntb   = blockIdx.y * NT;

    const _Float16* ap = A + (((size_t)blockIdx.x * Kc) * 32 + lane) * 16;
    const _Float16* bp = Bp + ((size_t)ntb * 32 + lane) * 16;
    const size_t bstep = (size_t)(Np >> 4) * 512;   // halves per K-chunk of B

    v8f acc[NT];
#pragma unroll
    for (int t = 0; t < NT; ++t) acc[t] = v8f{};

#pragma unroll 2
    for (int kc = 0; kc < Kc; ++kc) {
        v16h av = *(const v16h*)ap;
        if (kc + 1 < Kc) __builtin_prefetch(ap + 512, 0, 1);
#pragma unroll
        for (int t = 0; t < NT; ++t) {
            v16h bv = *(const v16h*)(bp + (size_t)t * 512);
            acc[t] = __builtin_amdgcn_wmma_f32_16x16x32_f16(
                false, av, false, bv, (short)0, acc[t], false, false);
        }
        ap += 512;
        bp += bstep;
    }

    const float rs = rsqrtf(1.0f + 1e-5f);
#pragma unroll
    for (int t = 0; t < NT; ++t) {
        int n = (ntb + t) * 16 + lr;
        if (n < Cout) {
            float bval = bias[n];
            float sc = gsc[n] * rs;
            float sh = bsh[n];
#pragma unroll
            for (int r = 0; r < 8; ++r) {
                int mr = mBase + r + hsel * 8;
                if (mr < Mtot) {
                    float v = acc[t][r] + bval;
                    v = (v > 0.0f) ? v : 0.01f * v;
                    v = v * sc + sh;
                    int b2 = mr / (OH * OW); int rr = mr % (OH * OW);
                    int yy = rr / OW, xx = rr % OW;
                    size_t oidx = (((size_t)b2 * Cout + n) * OH + yy) * OW + xx;
                    if (residual) v += residual[oidx];
                    out[oidx] = v;
                }
            }
        }
    }
}

// ---------------------------------------------------------------------------
// Dynamic-filter deform stage (LDS-staged frame; d3 streamed from L2).
// ---------------------------------------------------------------------------
__global__ void __launch_bounds__(256) deform_stage(
    const float* __restrict__ d3, const float* __restrict__ frame,
    float* __restrict__ outslice)
{
    __shared__ float img[4096];
    const int b = blockIdx.x >> 4;
    const int pixBase = (blockIdx.x & 15) * 256;
    const float* fb = frame + (size_t)b * 4096;
    for (int t = threadIdx.x; t < 4096; t += 256) img[t] = fb[t];
    __syncthreads();

    const int p = pixBase + threadIdx.x;
    const int y = p >> 6, x = p & 63;
    const int y57 = (y * 57) >> 6;
    const int x57 = (x * 57) >> 6;
    const float* dv = d3 + (size_t)b * 243 * 3249 + y57 * 57 + x57;

    float mx = -1e30f;
    for (int c = 0; c < 243; ++c) mx = fmaxf(mx, dv[c * 3249]);
    float s = 0.0f;
    for (int c = 0; c < 243; ++c) s += __expf(dv[c * 3249] - mx);
    const float inv = 1.0f / s;

    float pred = 0.0f;
    for (int k = 0; k < 81; ++k) {
        float fc  = __expf(dv[k * 3249]         - mx) * inv;
        float fox = __expf(dv[(81 + k) * 3249]  - mx) * inv;
        float foy = __expf(dv[(162 + k) * 3249] - mx) * inv;
        int ky = k / 9, kx = k % 9;
        int py = y - 4 + ky, px = x - 4 + kx;
        float pv = (py >= 0 && py < 64 && px >= 0 && px < 64) ? img[py * 64 + px] : 0.0f;
        float ys = (float)y + (float)(ky - 4) + pv * foy;
        float xs = (float)x + (float)(kx - 4) + pv * fox;
        float y0 = floorf(ys), x0 = floorf(xs);
        float wy = ys - y0, wx = xs - x0;
        int y0i = (int)y0, x0i = (int)x0;
        float v00 = (y0i >= 0 && y0i < 64 && x0i >= 0 && x0i < 64) ? img[y0i * 64 + x0i] : 0.0f;
        float v01 = (y0i >= 0 && y0i < 64 && x0i + 1 < 64 && x0i + 1 >= 0) ? img[y0i * 64 + x0i + 1] : 0.0f;
        float v10 = (y0i + 1 < 64 && y0i + 1 >= 0 && x0i >= 0 && x0i < 64) ? img[(y0i + 1) * 64 + x0i] : 0.0f;
        float v11 = (y0i + 1 < 64 && y0i + 1 >= 0 && x0i + 1 < 64 && x0i + 1 >= 0) ? img[(y0i + 1) * 64 + x0i + 1] : 0.0f;
        float samp = v00 * (1 - wy) * (1 - wx) + v01 * (1 - wy) * wx
                   + v10 * wy * (1 - wx)       + v11 * wy * wx;
        pred += samp * fc;
    }
    outslice[(size_t)b * 4096 + p] = pred;
}

// ---------------------------------------------------------------------------
// host driver
// ---------------------------------------------------------------------------
extern "C" void kernel_launch(void* const* d_in, const int* in_sizes, int n_in,
                              void* d_out, int out_size, void* d_ws, size_t ws_size,
                              hipStream_t stream) {
    (void)in_sizes; (void)out_size; (void)ws_size;
    const float* input = (const float*)d_in[0];
    const float* P[33];
    for (int i = 1; i < 33 && i < n_in; ++i) P[i] = (const float*)d_in[i];
    float* out = (float*)d_out;

    uint8_t* base = (uint8_t*)d_ws;
    size_t off = 0;
    auto alloc = [&](size_t bytes) -> void* {
        void* p = base + off;
        off = (off + bytes + 255) & ~(size_t)255;
        return p;
    };
    // B-fragment weight packs (halves): Kc*(Np/16)*512
    _Float16* wpEn1 = (_Float16*)alloc((size_t)3  * 2  * 512 * 2);
    _Float16* wpEn2 = (_Float16*)alloc((size_t)81 * 4  * 512 * 2);
    _Float16* wpEn3 = (_Float16*)alloc((size_t)18 * 8  * 512 * 2);
    _Float16* wpH1  = (_Float16*)alloc((size_t)36 * 8  * 512 * 2);
    _Float16* wpH2  = (_Float16*)alloc((size_t)36 * 8  * 512 * 2);
    _Float16* wpDe1 = (_Float16*)alloc((size_t)36 * 4  * 512 * 2);
    _Float16* wpDe2 = (_Float16*)alloc((size_t)18 * 2  * 512 * 2);
    _Float16* wpDe3 = (_Float16*)alloc((size_t)9  * 16 * 512 * 2);
    // shared A-fragment buffer: max over layers = de3 (3249*9*512 halves)
    _Float16* Afrag = (_Float16*)alloc((size_t)3249 * 9 * 512 * 2);
    float* frame = (float*)alloc((size_t)65536    * 4);
    float* a1    = (float*)alloc((size_t)524288   * 4);
    float* a2    = (float*)alloc((size_t)262144   * 4);
    float* a3    = (float*)alloc((size_t)131072   * 4);
    float* hb    = (float*)alloc((size_t)131072   * 4);
    float* hid   = (float*)alloc((size_t)131072   * 4);
    float* dd1   = (float*)alloc((size_t)230400   * 4);
    float* dd2   = (float*)alloc((size_t)430592   * 4);
    float* dd3   = (float*)alloc((size_t)12632112 * 4);

    zero_f32<<<(131072 + 255) / 256, 256, 0, stream>>>(hid, 131072);

    auto pack = [&](const float* w, _Float16* wp, int Kc, int Np, int Cin,
                    int Cout, int KH, int Ktot, int dec) {
        int nth = Kc * (Np >> 4) * 32;
        pack_w_frag<<<(nth + 255) / 256, 256, 0, stream>>>(w, wp, Kc, Np, Cin,
                                                           Cout, KH, Ktot, dec);
    };
    pack(P[1],  wpEn1, 3,  32,  1,   32,  9, 81,   0);
    pack(P[5],  wpEn2, 81, 64,  32,  64,  9, 2592, 0);
    pack(P[9],  wpEn3, 18, 128, 64,  128, 3, 576,  0);
    pack(P[13], wpH1,  36, 128, 128, 128, 3, 1152, 0);
    pack(P[17], wpH2,  36, 128, 128, 128, 3, 1152, 0);
    pack(P[21], wpDe1, 36, 64,  128, 64,  3, 1152, 1);
    pack(P[25], wpDe2, 18, 32,  64,  32,  3, 576,  1);
    pack(P[29], wpDe3, 9,  256, 32,  243, 3, 288,  1);

    auto conv = [&](const float* in, const _Float16* wp, const float* bias,
                    const float* g, const float* bb, const float* res, float* o,
                    int B, int Cin, int IH, int IW, int Cout, int OH, int OW,
                    int KH, int st, int pd, int Ktot, int Kc, int Np, int dec) {
        int Mtot = B * OH * OW;
        int Mtiles = (Mtot + 15) / 16;
        int nth = Mtiles * Kc * 32;
        build_a<<<(nth + 255) / 256, 256, 0, stream>>>(
            in, Afrag, Mtiles, Kc, B, Cin, IH, IW, OH, OW, KH, st, pd, Ktot, dec);
        if (Np == 32) {
            dim3 grid(Mtiles, 1);
            wmma_gemm<2><<<grid, 32, 0, stream>>>(Afrag, wp, bias, g, bb, res, o,
                                                  Kc, Np, Mtot, Cout, OH, OW);
        } else {
            dim3 grid(Mtiles, Np / 64);
            wmma_gemm<4><<<grid, 32, 0, stream>>>(Afrag, wp, bias, g, bb, res, o,
                                                  Kc, Np, Mtot, Cout, OH, OW);
        }
    };

    for (int i = 0; i < 19; ++i) {
        const float* prev = (i >= 10) ? (out + (size_t)(i - 10) * 65536) : out;
        pack_frame<<<256, 256, 0, stream>>>(input, prev, frame, i, 65536);

        conv(frame, wpEn1, P[2],  P[3],  P[4],  nullptr, a1,
             16, 1,   64, 64, 32,  32, 32, 9, 2, 4, 81,   3,  32,  0);
        conv(a1,    wpEn2, P[6],  P[7],  P[8],  nullptr, a2,
             16, 32,  32, 32, 64,  16, 16, 9, 2, 4, 2592, 81, 64,  0);
        conv(a2,    wpEn3, P[10], P[11], P[12], nullptr, a3,
             16, 64,  16, 16, 128, 8,  8,  3, 2, 1, 576,  18, 128, 0);
        conv(a3,    wpH1,  P[14], P[15], P[16], nullptr, hb,
             16, 128, 8,  8,  128, 8,  8,  3, 1, 1, 1152, 36, 128, 0);
        conv(hb,    wpH2,  P[18], P[19], P[20], hid,     hid,
             16, 128, 8,  8,  128, 8,  8,  3, 1, 1, 1152, 36, 128, 0);
        conv(hid,   wpDe1, P[22], P[23], P[24], nullptr, dd1,
             16, 128, 8,  8,  64,  15, 15, 3, 1, 1, 1152, 36, 64,  1);
        conv(dd1,   wpDe2, P[26], P[27], P[28], nullptr, dd2,
             16, 64,  15, 15, 32,  29, 29, 3, 1, 1, 576,  18, 32,  1);
        conv(dd2,   wpDe3, P[30], P[31], P[32], nullptr, dd3,
             16, 32,  29, 29, 243, 57, 57, 3, 1, 1, 288,  9,  256, 1);

        if (i >= 9) {
            deform_stage<<<16 * 16, 256, 0, stream>>>(
                dd3, frame, out + (size_t)(i - 9) * 65536);
        }
    }
}
